// SRNN_70050916598281
// MI455X (gfx1250) — matrix operand order repaired and checked by reading, not compiled, not run on hardware
//
#include <hip/hip_runtime.h>
#include <hip/hip_bf16.h>

typedef __attribute__((ext_vector_type(16))) __bf16 v16bf;
typedef __attribute__((ext_vector_type(8)))  float  v8f;

#define NN 128
#define TSTEPS 20
#define MSPAT (128*127)

static __device__ __forceinline__ unsigned short f2bf(float f){
    unsigned int u = __float_as_uint(f);
    u += 0x7FFFu + ((u >> 16) & 1u);          // round-to-nearest-even
    return (unsigned short)(u >> 16);
}
static __device__ __forceinline__ float sigf(float x){
    return 1.0f / (1.0f + __expf(-x));
}

union Frag { v16bf v; unsigned int u[8]; };

// A: 16x32 bf16 tile, row-major source, leading dim ld (elements).
// Lane = M; VGPR v holds K pair; hi half-wave K+8. Two 16B loads per fragment.
static __device__ __forceinline__ void loadA(Frag& a, const unsigned short* p,
                                             int ld, int kBase, int lane){
    int m = lane & 15, hi = (lane >> 4) & 1;
    const uint4* q = (const uint4*)(p + m*ld + kBase + hi*8);
    uint4 q0 = q[0];   // K pairs 0,2,4,6   (+hi*8)
    uint4 q1 = q[2];   // K pairs 16,18,20,22
    a.u[0]=q0.x; a.u[1]=q0.y; a.u[2]=q0.z; a.u[3]=q0.w;
    a.u[4]=q1.x; a.u[5]=q1.y; a.u[6]=q1.z; a.u[7]=q1.w;
}

// B: 32x16 bf16 tile from fragment-order swizzled weights:
// P[((chunk*numTiles + tile)*32 + lane)*8 + v]  (dwords, 32B per lane, coalesced)
static __device__ __forceinline__ void loadB(Frag& b, const unsigned int* P,
                                             int numTiles, int chunk, int tile, int lane){
    const uint4* q = (const uint4*)(P + (size_t)((chunk*numTiles + tile)*32 + lane)*8);
    uint4 q0 = q[0], q1 = q[1];
    b.u[0]=q0.x; b.u[1]=q0.y; b.u[2]=q0.z; b.u[3]=q0.w;
    b.u[4]=q1.x; b.u[5]=q1.y; b.u[6]=q1.z; b.u[7]=q1.w;
}

static __device__ __forceinline__ v8f wmma_bf16(const Frag& a, const Frag& b, v8f c){
    return __builtin_amdgcn_wmma_f32_16x16x32_bf16(false, a.v, false, b.v,
                                                   (short)0, c, false, false);
}

// ---------------------------------------------------------------------------
// Fused edge LSTM step:
//   g = relu(edge@enc_w.T+enc_b)@wih.T + h@whh.T + (bih+bhh); LSTM update.
// Wave = 16 rows x 64 hidden units x 4 gates (16 accumulators, 160 WMMA).
// Grid: (M/16)*4 waves, 8 waves/block (4 hid-waves of a row-tile share a block).
// ---------------------------------------------------------------------------
__global__ void __launch_bounds__(256, 1)
edge_lstm_step(const float* __restrict__ edges_t, int diag,
    const float* __restrict__ enc_w, const float* __restrict__ enc_b,
    const unsigned int* __restrict__ wihP,   // swizzled [2][64 tiles][32][8]
    const unsigned int* __restrict__ whhP,   // swizzled [8][64 tiles][32][8]
    const float* __restrict__ biasC,         // [1024] = bih + bhh
    unsigned short* __restrict__ hB,         // [M][256] bf16 state (in/out)
    float* __restrict__ hF,                  // [M][256] f32 h (out)
    float* __restrict__ cF)                  // [M][256] f32 c (in/out)
{
    __shared__ __align__(16) unsigned short sEnc[8 * 16 * 64];
    int lane = threadIdx.x & 31;
    int wInB = threadIdx.x >> 5;
    int wid  = blockIdx.x * 8 + wInB;
    int rowBase = (wid >> 2) * 16;
    int hb4     = (wid & 3) * 4;           // hidden-tile base in 16-col units
    int hBase   = hb4 * 16;
    unsigned short* myEnc = sEnc + wInB * 1024;

    // encoder: relu(e @ enc_w.T + enc_b) -> bf16 tile [16][64] in LDS
    for (int it = 0; it < 32; ++it){
        int idx = it*32 + lane;
        int m = idx >> 6, k = idx & 63;
        int r = rowBase + m;
        int eidx;
        if (diag) eidx = r * 129;
        else { int i = r / 127, jj = r - i*127; int j = jj + (jj >= i ? 1 : 0); eidx = i*128 + j; }
        float e0 = edges_t[eidx*2], e1 = edges_t[eidx*2 + 1];
        float v = fmaf(e0, enc_w[2*k], fmaf(e1, enc_w[2*k+1], enc_b[k]));
        myEnc[m*64 + k] = f2bf(v > 0.f ? v : 0.f);
    }
    __syncthreads();

    v8f acc[16];
    #pragma unroll
    for (int i = 0; i < 16; ++i)
        #pragma unroll
        for (int r = 0; r < 8; ++r) acc[i][r] = 0.f;

    // x @ wih.T   (K = 64, A from LDS)
    #pragma unroll
    for (int c = 0; c < 2; ++c){
        Frag a; loadA(a, myEnc, 64, c*32, lane);
        #pragma unroll
        for (int g = 0; g < 4; ++g)
            #pragma unroll
            for (int j = 0; j < 4; ++j){
                Frag b; loadB(b, wihP, 64, c, g*16 + hb4 + j, lane);
                acc[g*4 + j] = wmma_bf16(a, b, acc[g*4 + j]);
            }
    }
    // h @ whh.T   (K = 256, A from global bf16 state)
    const unsigned short* hRow = hB + (size_t)rowBase * 256;
    for (int c = 0; c < 8; ++c){
        Frag a; loadA(a, hRow, 256, c*32, lane);
        #pragma unroll
        for (int g = 0; g < 4; ++g)
            #pragma unroll
            for (int j = 0; j < 4; ++j){
                Frag b; loadB(b, whhP, 64, c, g*16 + hb4 + j, lane);
                acc[g*4 + j] = wmma_bf16(a, b, acc[g*4 + j]);
            }
    }
    __syncthreads();   // all hid-waves of this row-tile done reading hB

    // LSTM nonlinearity + in-place state update
    int m0 = ((lane >> 4) & 1) * 8;
    int n  = lane & 15;
    #pragma unroll
    for (int j = 0; j < 4; ++j){
        int d = hBase + j*16 + n;
        float bI = biasC[d], bF2 = biasC[256 + d], bG = biasC[512 + d], bO = biasC[768 + d];
        #pragma unroll
        for (int r = 0; r < 8; ++r){
            int row = rowBase + m0 + r;
            float gi = acc[0*4 + j][r] + bI;
            float gf = acc[1*4 + j][r] + bF2;
            float gg = acc[2*4 + j][r] + bG;
            float go = acc[3*4 + j][r] + bO;
            float cOld = cF[(size_t)row*256 + d];
            float c2 = sigf(gf)*cOld + sigf(gi)*tanhf(gg);
            float hv = sigf(go)*tanhf(c2);
            cF[(size_t)row*256 + d] = c2;
            hF[(size_t)row*256 + d] = hv;
            hB[(size_t)row*256 + d] = f2bf(hv);
        }
    }
}

// ---------------------------------------------------------------------------
// out[M,64] = A_bf16[M,256] @ Bp + bias   (te / se attention projections)
// Wave = 16 rows x 64 cols; grid = M/16 waves, 8/block.
// ---------------------------------------------------------------------------
__global__ void __launch_bounds__(256, 1)
gemm_bias64(const unsigned short* __restrict__ A,
    const unsigned int* __restrict__ BP,      // swizzled [8][4 tiles][32][8]
    const float* __restrict__ bias, float* __restrict__ out)
{
    int lane = threadIdx.x & 31;
    int wid  = blockIdx.x * 8 + (threadIdx.x >> 5);
    int rowBase = wid * 16;
    v8f acc[4];
    #pragma unroll
    for (int j = 0; j < 4; ++j)
        #pragma unroll
        for (int r = 0; r < 8; ++r) acc[j][r] = 0.f;
    const unsigned short* aRow = A + (size_t)rowBase * 256;
    for (int c = 0; c < 8; ++c){
        Frag a; loadA(a, aRow, 256, c*32, lane);
        #pragma unroll
        for (int j = 0; j < 4; ++j){
            Frag b; loadB(b, BP, 4, c, j, lane);
            acc[j] = wmma_bf16(a, b, acc[j]);
        }
    }
    int m0 = ((lane >> 4) & 1) * 8, n = lane & 15;
    #pragma unroll
    for (int j = 0; j < 4; ++j){
        int col = j*16 + n; float bv = bias[col];
        #pragma unroll
        for (int r = 0; r < 8; ++r)
            out[(size_t)(rowBase + m0 + r)*64 + col] = acc[j][r] + bv;
    }
}

// ---------------------------------------------------------------------------
// Per-node attention: logits = (se . te)*TEMP, softmax over 127, weighted sum.
// One block (128 threads) per node.
// ---------------------------------------------------------------------------
__global__ void attn_combine(const float* __restrict__ se, const float* __restrict__ te,
    const float* __restrict__ hsF, float* __restrict__ weighted)
{
    __shared__ float sTe[64], sL[128], sR[128], sW[128];
    int i = blockIdx.x, tid = threadIdx.x;
    if (tid < 64) sTe[tid] = te[i*64 + tid];
    __syncthreads();
    float lg = -1e30f;
    if (tid < 127){
        const float* sp = se + (size_t)(i*127 + tid)*64;
        float d = 0.f;
        for (int a = 0; a < 64; ++a) d = fmaf(sp[a], sTe[a], d);
        lg = d * 15.875f;   // (N-1)/sqrt(64)
    }
    sL[tid] = lg; sR[tid] = lg;
    __syncthreads();
    for (int s = 64; s > 0; s >>= 1){
        if (tid < s) sR[tid] = fmaxf(sR[tid], sR[tid + s]);
        __syncthreads();
    }
    float mx = sR[0];
    __syncthreads();
    float e = (tid < 127) ? __expf(sL[tid] - mx) : 0.f;
    sR[tid] = e;
    __syncthreads();
    for (int s = 64; s > 0; s >>= 1){
        if (tid < s) sR[tid] += sR[tid + s];
        __syncthreads();
    }
    float ssum = sR[0];
    sW[tid] = e / ssum;
    __syncthreads();
    float a0 = 0.f, a1 = 0.f;
    for (int k = 0; k < 127; ++k){
        float wk = sW[k];
        const float* hp = hsF + (size_t)(i*127 + k)*256;
        a0 = fmaf(wk, hp[tid],        a0);
        a1 = fmaf(wk, hp[tid + 128],  a1);
    }
    weighted[i*256 + tid]       = a0;
    weighted[i*256 + tid + 128] = a1;
}

// ---------------------------------------------------------------------------
// Node LSTM + output head. One block (512 threads) per node.
// ---------------------------------------------------------------------------
__global__ void __launch_bounds__(512)
node_step(const float* __restrict__ nodes_t,
    const float* __restrict__ n_enc_w, const float* __restrict__ n_enc_b,
    const float* __restrict__ n_attn_w, const float* __restrict__ n_attn_b,
    const float* __restrict__ n_wih, const float* __restrict__ n_whh,
    const float* __restrict__ n_bih, const float* __restrict__ n_bhh,
    const float* __restrict__ out_w, const float* __restrict__ out_b,
    const float* __restrict__ htF, const float* __restrict__ weighted,
    float* __restrict__ hN, float* __restrict__ cN, float* __restrict__ outT)
{
    __shared__ float sx[128], sg[512], sh[128];
    int i = blockIdx.x, tid = threadIdx.x;
    if (tid < 64){
        float v = fmaf(nodes_t[i*2], n_enc_w[2*tid],
                  fmaf(nodes_t[i*2+1], n_enc_w[2*tid+1], n_enc_b[tid]));
        sx[tid] = v > 0.f ? v : 0.f;
    } else if (tid < 128){
        int q = tid - 64;
        const float* wr = n_attn_w + q*512;
        float acc = n_attn_b[q];
        for (int k = 0; k < 256; ++k) acc = fmaf(htF[i*256 + k],      wr[k],       acc);
        for (int k = 0; k < 256; ++k) acc = fmaf(weighted[i*256 + k], wr[256 + k], acc);
        sx[tid] = acc > 0.f ? acc : 0.f;
    }
    __syncthreads();
    {
        int g = tid;
        const float* w1 = n_wih + g*128;
        const float* w2 = n_whh + g*128;
        float acc = n_bih[g] + n_bhh[g];
        for (int k = 0; k < 128; ++k) acc = fmaf(sx[k], w1[k], acc);
        for (int k = 0; k < 128; ++k) acc = fmaf(hN[i*128 + k], w2[k], acc);
        sg[g] = acc;
    }
    __syncthreads();
    if (tid < 128){
        float gi = sigf(sg[tid]), gf = sigf(sg[128 + tid]);
        float gg = tanhf(sg[256 + tid]), go = sigf(sg[384 + tid]);
        float c2 = gf * cN[i*128 + tid] + gi * gg;
        float hv = go * tanhf(c2);
        cN[i*128 + tid] = c2; hN[i*128 + tid] = hv; sh[tid] = hv;
    }
    __syncthreads();
    if (tid < 5){
        const float* wr = out_w + tid*128;
        float acc = out_b[tid];
        for (int k = 0; k < 128; ++k) acc = fmaf(sh[k], wr[k], acc);
        outT[i*5 + tid] = acc;
    }
}

// ---------------------------------------------------------------------------
// Prep kernels: swizzle weights W[OUT,IN] (fp32) into WMMA B-fragment order.
// dword idx = ((chunk*(OUT/16) + tile)*32 + lane)*8 + v
//   k = chunk*32 + (lane>=16)*8 + (v>=4)*16 + 2(v&3) ; col = tile*16 + (lane&15)
// ---------------------------------------------------------------------------
__global__ void pack_swz(const float* __restrict__ W, unsigned int* __restrict__ P,
                         int OUT, int IN){
    int total = OUT * (IN >> 1);
    int tilesN = OUT >> 4;
    for (int idx = blockIdx.x*blockDim.x + threadIdx.x; idx < total;
         idx += gridDim.x*blockDim.x){
        int v    = idx & 7;
        int lane = (idx >> 3) & 31;
        int rest = idx >> 8;
        int t    = rest % tilesN;
        int c    = rest / tilesN;
        int n    = t*16 + (lane & 15);
        int k    = c*32 + (lane >> 4)*8 + (v >> 2)*16 + 2*(v & 3);
        unsigned int lo = f2bf(W[(size_t)n*IN + k]);
        unsigned int hh = f2bf(W[(size_t)n*IN + k + 1]);
        P[idx] = lo | (hh << 16);
    }
}
__global__ void add_bias(const float* a, const float* b, float* o, int n){
    int i = blockIdx.x*blockDim.x + threadIdx.x;
    if (i < n) o[i] = a[i] + b[i];
}
__global__ void zero_f32(float* p, long long n){
    for (long long i = blockIdx.x*(long long)blockDim.x + threadIdx.x; i < n;
         i += (long long)gridDim.x*blockDim.x) p[i] = 0.f;
}

// ---------------------------------------------------------------------------
extern "C" void kernel_launch(void* const* d_in, const int* in_sizes, int n_in,
                              void* d_out, int out_size, void* d_ws, size_t ws_size,
                              hipStream_t stream)
{
    (void)in_sizes; (void)n_in; (void)out_size; (void)ws_size;
    const float* nodes    = (const float*)d_in[0];
    const float* edges    = (const float*)d_in[1];
    const float* t_enc_w  = (const float*)d_in[6];
    const float* t_enc_b  = (const float*)d_in[7];
    const float* t_wih    = (const float*)d_in[8];
    const float* t_whh    = (const float*)d_in[9];
    const float* t_bih    = (const float*)d_in[10];
    const float* t_bhh    = (const float*)d_in[11];
    const float* s_enc_w  = (const float*)d_in[12];
    const float* s_enc_b  = (const float*)d_in[13];
    const float* s_wih    = (const float*)d_in[14];
    const float* s_whh    = (const float*)d_in[15];
    const float* s_bih    = (const float*)d_in[16];
    const float* s_bhh    = (const float*)d_in[17];
    const float* att_t_w  = (const float*)d_in[18];
    const float* att_t_b  = (const float*)d_in[19];
    const float* att_s_w  = (const float*)d_in[20];
    const float* att_s_b  = (const float*)d_in[21];
    const float* n_enc_w  = (const float*)d_in[22];
    const float* n_enc_b  = (const float*)d_in[23];
    const float* n_attn_w = (const float*)d_in[24];
    const float* n_attn_b = (const float*)d_in[25];
    const float* n_wih    = (const float*)d_in[26];
    const float* n_whh    = (const float*)d_in[27];
    const float* n_bih    = (const float*)d_in[28];
    const float* n_bhh    = (const float*)d_in[29];
    const float* out_w    = (const float*)d_in[30];
    const float* out_b    = (const float*)d_in[31];

    char* ws = (char*)d_ws;
    size_t off = 0;
    auto alloc = [&](size_t bytes)->char*{
        char* p = ws + off; off = (off + bytes + 255) & ~(size_t)255; return p;
    };
    // --- state (zeroed every call; contiguous prefix) ---
    unsigned short* hsB = (unsigned short*)alloc((size_t)MSPAT*256*2);
    unsigned short* htB = (unsigned short*)alloc((size_t)NN*256*2);
    float* hsF = (float*)alloc((size_t)MSPAT*256*4);
    float* htF = (float*)alloc((size_t)NN*256*4);
    float* cS  = (float*)alloc((size_t)MSPAT*256*4);
    float* cT  = (float*)alloc((size_t)NN*256*4);
    float* hN  = (float*)alloc((size_t)NN*128*4);
    float* cN  = (float*)alloc((size_t)NN*128*4);
    size_t stateBytes = off;
    // --- scratch / packed weights ---
    float* se  = (float*)alloc((size_t)MSPAT*64*4);
    float* te  = (float*)alloc((size_t)NN*64*4);
    float* wtd = (float*)alloc((size_t)NN*256*4);
    unsigned int* t_wihP = (unsigned int*)alloc((size_t)32*1024*4);
    unsigned int* t_whhP = (unsigned int*)alloc((size_t)128*1024*4);
    unsigned int* s_wihP = (unsigned int*)alloc((size_t)32*1024*4);
    unsigned int* s_whhP = (unsigned int*)alloc((size_t)128*1024*4);
    unsigned int* attTP  = (unsigned int*)alloc((size_t)128*64*4);
    unsigned int* attSP  = (unsigned int*)alloc((size_t)128*64*4);
    float* biasT = (float*)alloc(1024*4);
    float* biasS = (float*)alloc(1024*4);

    zero_f32<<<2048, 256, 0, stream>>>((float*)ws, (long long)(stateBytes/4));
    pack_swz<<<128, 256, 0, stream>>>(t_wih, t_wihP, 1024, 64);
    pack_swz<<<512, 256, 0, stream>>>(t_whh, t_whhP, 1024, 256);
    pack_swz<<<128, 256, 0, stream>>>(s_wih, s_wihP, 1024, 64);
    pack_swz<<<512, 256, 0, stream>>>(s_whh, s_whhP, 1024, 256);
    pack_swz<<<32, 256, 0, stream>>>(att_t_w, attTP, 64, 256);
    pack_swz<<<32, 256, 0, stream>>>(att_s_w, attSP, 64, 256);
    add_bias<<<4, 256, 0, stream>>>(t_bih, t_bhh, biasT, 1024);
    add_bias<<<4, 256, 0, stream>>>(s_bih, s_bhh, biasS, 1024);

    for (int t = 0; t < TSTEPS; ++t){
        const float* edges_t = edges + (size_t)t * NN * NN * 2;
        // temporal edges: M=128 -> 8 row-tiles * 4 hid-tiles = 32 waves = 4 blocks
        edge_lstm_step<<<4, 256, 0, stream>>>(edges_t, 1, t_enc_w, t_enc_b,
            t_wihP, t_whhP, biasT, htB, htF, cT);
        // spatial edges: M=16256 -> 1016*4 = 4064 waves = 508 blocks
        edge_lstm_step<<<508, 256, 0, stream>>>(edges_t, 0, s_enc_w, s_enc_b,
            s_wihP, s_whhP, biasS, hsB, hsF, cS);
        // te: 8 waves = 1 block; se: 1016 waves = 127 blocks
        gemm_bias64<<<1, 256, 0, stream>>>(htB, attTP, att_t_b, te);
        gemm_bias64<<<127, 256, 0, stream>>>(hsB, attSP, att_s_b, se);
        attn_combine<<<NN, 128, 0, stream>>>(se, te, hsF, wtd);
        node_step<<<NN, 512, 0, stream>>>(nodes + (size_t)t*NN*2,
            n_enc_w, n_enc_b, n_attn_w, n_attn_b, n_wih, n_whh, n_bih, n_bhh,
            out_w, out_b, htF, wtd, hN, cN, (float*)d_out + (size_t)t*NN*5);
    }
}
